// StyleAttnProcessor_214748365488
// MI455X (gfx1250) — matrix operand order repaired and checked by reading, not compiled, not run on hardware
//
#include <hip/hip_runtime.h>

// ---------------------------------------------------------------------------
// Types
// ---------------------------------------------------------------------------
typedef unsigned short u16;
typedef __attribute__((ext_vector_type(4)))  unsigned short u16x4;
typedef __attribute__((ext_vector_type(8)))  unsigned short u16x8;
typedef __attribute__((ext_vector_type(16))) unsigned short u16x16;
typedef __attribute__((ext_vector_type(16))) __bf16        bf16x16;
typedef __attribute__((ext_vector_type(8)))  float          f32x8;
typedef __attribute__((ext_vector_type(4)))  float          f32x4;

// f32 -> bf16 round-to-nearest-even
__device__ __forceinline__ u16 f2bf(float f) {
  unsigned int u = __builtin_bit_cast(unsigned int, f);
  u += 0x7fffu + ((u >> 16) & 1u);
  return (u16)(u >> 16);
}

__device__ __forceinline__ bf16x16 make_frag(u16x8 lo, u16x8 hi) {
  u16x16 u = __builtin_shufflevector(lo, hi, 0,1,2,3,4,5,6,7,8,9,10,11,12,13,14,15);
  union { u16x16 u; bf16x16 b; } cv; cv.u = u; return cv.b;
}
// A-operand fragment (16-bit A 16x32 layout): lane holds K = kbase+{0..7} and kbase+16+{0..7}
__device__ __forceinline__ bf16x16 frag_ileave(const u16* rowbase, int kbase) {
  u16x8 lo = *(const u16x8*)(rowbase + kbase);
  u16x8 hi = *(const u16x8*)(rowbase + kbase + 16);
  return make_frag(lo, hi);
}
// B-operand fragment: lane holds 16 contiguous K values starting at p
__device__ __forceinline__ bf16x16 frag_contig(const u16* p) {
  u16x8 lo = *(const u16x8*)(p);
  u16x8 hi = *(const u16x8*)(p + 8);
  return make_frag(lo, hi);
}
__device__ __forceinline__ f32x8 wmma_bf16(bf16x16 a, bf16x16 b, f32x8 c) {
  return __builtin_amdgcn_wmma_f32_16x16x32_bf16(false, a, false, b, (short)0, c, false, false);
}

// CDNA5 async memory->LDS copy, 16B per lane; tracked by ASYNCcnt (ISA ch.10 / §08)
__device__ __forceinline__ void async_g2l_b128(unsigned lds_off, const void* gptr) {
  asm volatile("global_load_async_to_lds_b128 %0, %1, off"
               :: "v"(lds_off), "v"(gptr) : "memory");
}
__device__ __forceinline__ void wait_async0() {
  asm volatile("s_wait_asynccnt 0x0" ::: "memory");
}
__device__ __forceinline__ unsigned lds_off_u32(const void* p) {
  // generic (flat) LDS pointer -> 32-bit LDS address (low 32 bits of the address)
  return (unsigned)(unsigned long long)p;
}

// ---------------------------------------------------------------------------
// Problem constants
// ---------------------------------------------------------------------------
#define NB    8
#define SQ    4096
#define DM    1280
#define SE    77
#define SS    4
#define CI    768
#define NH    20
#define DH    64
#define SKV   96          // 81 valid keys padded to 96
#define SVALID 81

// ---------------------------------------------------------------------------
// Kernel 1: tiny K/V (+style) projections -> bf16, padded + V transposed
//   Kcat [B,H,96,64] bf16 ; Vt [B,H,64,96] bf16
// ---------------------------------------------------------------------------
__global__ __launch_bounds__(256) void kv_prep_kernel(
    const float* __restrict__ enc, const float* __restrict__ sty,
    const float* __restrict__ Wk,  const float* __restrict__ Wv,
    const float* __restrict__ Wks, const float* __restrict__ Wvs,
    u16* __restrict__ Kc, u16* __restrict__ Vt)
{
  int idx = blockIdx.x * 256 + threadIdx.x;           // over B*96*1280
  if (idx >= NB * SKV * DM) return;
  int n = idx % DM;
  int j = (idx / DM) % SKV;
  int b = idx / (DM * SKV);
  float kv = 0.f, vv = 0.f;
  if (j < SE) {
    const float* e = enc + ((size_t)b * SE + j) * CI;
    for (int c = 0; c < CI; ++c) {
      float ec = e[c];
      kv += ec * Wk[(size_t)c * DM + n];
      vv += ec * Wv[(size_t)c * DM + n];
    }
  } else if (j < SVALID) {
    const float* e = sty + ((size_t)b * SS + (j - SE)) * CI;
    for (int c = 0; c < CI; ++c) {
      float ec = e[c];
      kv += ec * Wks[(size_t)c * DM + n];   // STYLE_SCALE == 1.0
      vv += ec * Wvs[(size_t)c * DM + n];
    }
  }
  int h = n >> 6, d = n & 63;
  size_t bh = (size_t)b * NH + h;
  Kc[(bh * SKV + j) * DH + d] = f2bf(kv);
  Vt[(bh * DH + d) * SKV + j] = f2bf(vv);
}

// ---------------------------------------------------------------------------
// Kernel 2: weight prep, done once: W [K,N] f32 -> Wt [N,K] bf16 (transposed)
// ---------------------------------------------------------------------------
__global__ __launch_bounds__(256) void wt_prep_kernel(
    const float* __restrict__ W, u16* __restrict__ Wt, int N, int K)
{
  int idx = blockIdx.x * 256 + threadIdx.x;
  if (idx >= N * K) return;
  int n = idx % N;                  // consecutive threads read W row-contiguous
  int k = idx / N;
  Wt[(size_t)n * K + k] = f2bf(W[(size_t)k * N + n]);
}

// ---------------------------------------------------------------------------
// Kernel 3/5: double-buffered bf16 WMMA GEMM  C[M,N] = A[M,K] * Bt[N,K]^T (+bias)
//   Bt is pre-transposed bf16 weight -> staged via GLOBAL_LOAD_ASYNC_TO_LDS_B128.
//   A: bf16 (async staged) or f32 (global loads issued pre-compute, converted
//   to LDS post-compute, so latency hides under the WMMAs).
//   256 threads = 8 waves, 4(M) x 2(N); wave tile 32x64; block tile 128x128x32
// ---------------------------------------------------------------------------
template<int A_BF16, int OUT_BF16, int ADD_BIAS>
__global__ __launch_bounds__(256) void gemm_wmma_kernel(
    const void* __restrict__ Ap, const u16* __restrict__ Btp,
    void* __restrict__ Cp, const float* __restrict__ bias,
    int M, int N, int K)
{
  constexpr int BM = 128, BN = 128, BK = 32, LDT = 48;  // LDS row stride (16B multiple)
  __shared__ u16 As[2][BM * LDT];   // [m][k], double buffered (12KB each)
  __shared__ u16 Bs[2][BN * LDT];   // [n][k], double buffered

  const int tid  = threadIdx.x;
  const int lane = tid & 31, wave = tid >> 5;
  const int la   = lane & 15, half = lane >> 4;
  const int wm   = wave & 3,  wn   = wave >> 2;      // 4x2 wave grid
  const int m0   = blockIdx.y * BM, n0 = blockIdx.x * BN;

  const unsigned aL[2] = { lds_off_u32(&As[0][0]), lds_off_u32(&As[1][0]) };
  const unsigned bL[2] = { lds_off_u32(&Bs[0][0]), lds_off_u32(&Bs[1][0]) };

  f32x8 acc[2][4];
  for (int mi = 0; mi < 2; ++mi)
    for (int ni = 0; ni < 4; ++ni)
      for (int e = 0; e < 8; ++e) acc[mi][ni][e] = 0.f;

  const float* Af = (const float*)Ap;
  const u16*   Ab = (const u16*)Ap;
  f32x4 areg[4];

  // --- staging helpers --------------------------------------------------
  auto stageB_async = [&](int k0, int buf) {
    for (int i = 0; i < 2; ++i) {
      int c = tid + i * 256;          // 512 x 16B chunks (128 rows x 4)
      int r = c >> 2, q = c & 3;
      async_g2l_b128(bL[buf] + (unsigned)(r * (LDT * 2) + q * 16),
                     Btp + (size_t)(n0 + r) * K + k0 + q * 8);
    }
  };
  auto stageA_async = [&](int k0, int buf) {
    for (int i = 0; i < 2; ++i) {
      int c = tid + i * 256;
      int r = c >> 2, q = c & 3;
      async_g2l_b128(aL[buf] + (unsigned)(r * (LDT * 2) + q * 16),
                     Ab + (size_t)(m0 + r) * K + k0 + q * 8);
    }
  };
  auto loadA_f32 = [&](int k0) {
    for (int i = 0; i < 4; ++i) {
      int c = tid + i * 256;          // 1024 f32x4 chunks
      int r = c >> 3, kq = (c & 7) * 4;
      areg[i] = *(const f32x4*)(Af + (size_t)(m0 + r) * K + k0 + kq);
    }
  };
  auto storeA_f32 = [&](int buf) {
    for (int i = 0; i < 4; ++i) {
      int c = tid + i * 256;
      int r = c >> 3, kq = (c & 7) * 4;
      u16x4 v; v.x = f2bf(areg[i].x); v.y = f2bf(areg[i].y);
               v.z = f2bf(areg[i].z); v.w = f2bf(areg[i].w);
      *(u16x4*)(&As[buf][r * LDT + kq]) = v;
    }
  };

  // --- prologue: stage tile 0 into buffer 0 ---
  if (A_BF16) stageA_async(0, 0);
  else        { loadA_f32(0); storeA_f32(0); }
  stageB_async(0, 0);
  wait_async0();
  __syncthreads();

  const int KT = K / BK;
  for (int kt = 0; kt < KT; ++kt) {
    const int cur = kt & 1, nxt = cur ^ 1;
    const bool has_next = (kt + 1 < KT);
    if (has_next) {                       // issue next tile (other buffer)
      const int k0n = (kt + 1) * BK;
      if (A_BF16) stageA_async(k0n, nxt);
      else        loadA_f32(k0n);         // in flight during compute
      stageB_async(k0n, nxt);
    }
    // compute current tile
    bf16x16 af[2], bfr[4];
    for (int mi = 0; mi < 2; ++mi)
      af[mi] = frag_contig(&As[cur][(wm * 32 + mi * 16 + la) * LDT + half * 16]);
    for (int ni = 0; ni < 4; ++ni)
      bfr[ni] = frag_contig(&Bs[cur][(wn * 64 + ni * 16 + la) * LDT + half * 16]);
    for (int mi = 0; mi < 2; ++mi)
      for (int ni = 0; ni < 4; ++ni)
        acc[mi][ni] = wmma_bf16(af[mi], bfr[ni], acc[mi][ni]);

    if (has_next && !A_BF16) storeA_f32(nxt);  // convert after compute
    wait_async0();                              // async writes for nxt landed
    __syncthreads();                            // publish; also fences ds stores
  }

  // --- epilogue: C/D layout = VGPR r -> row (half*8 + r), col = lane&15 ---
  for (int mi = 0; mi < 2; ++mi) {
    int row0 = m0 + wm * 32 + mi * 16 + half * 8;
    for (int ni = 0; ni < 4; ++ni) {
      int col = n0 + wn * 64 + ni * 16 + la;
      float bv = ADD_BIAS ? bias[col] : 0.f;
      for (int r = 0; r < 8; ++r) {
        float v = acc[mi][ni][r] + bv;
        size_t off = (size_t)(row0 + r) * N + col;
        if (OUT_BF16) ((u16*)Cp)[off] = f2bf(v);
        else          ((float*)Cp)[off] = v;
      }
    }
  }
}

// ---------------------------------------------------------------------------
// Kernel 4: attention core. One block = (b, h, 64-row q chunk).
//   128 threads = 4 waves, each wave owns 16 q rows.
//   K/V staged with async-to-LDS; S = Q*K^T (scaled), f32 softmax over 81
//   keys (pad masked), O = P*V.
// ---------------------------------------------------------------------------
__global__ __launch_bounds__(128) void attn_wmma_kernel(
    const u16* __restrict__ Q,   // [B,Sq,1280] bf16
    const u16* __restrict__ Kc,  // [B,H,96,64] bf16
    const u16* __restrict__ Vt,  // [B,H,64,96] bf16 (transposed)
    u16* __restrict__ O)         // [B,Sq,1280] bf16
{
  constexpr int CHUNK = 64;
  __shared__ u16   Ks[SKV * DH];        // 12 KB [key][d]
  __shared__ u16   Vs[DH * SKV];        // 12 KB [d][key]
  __shared__ float Sb[CHUNK * SKV];     // 24 KB scores (P bf16 packed in place)

  const int tid  = threadIdx.x;
  const int lane = tid & 31, wave = tid >> 5;
  const int la   = lane & 15, half = lane >> 4;
  const int bid  = blockIdx.x;
  const int qc   = bid & 63;            // 4096/64 chunks
  const int bh   = bid >> 6;            // b*H + h
  const int h    = bh % NH, b = bh / NH;

  // stage K / V^T for this (b,h) via async copies (16B/lane, ASYNCcnt)
  {
    const unsigned ksb = lds_off_u32(Ks), vsb = lds_off_u32(Vs);
    const u16* kg = Kc + (size_t)bh * SKV * DH;
    const u16* vg = Vt + (size_t)bh * DH * SKV;
    for (int i = tid; i < SKV * DH / 8; i += 128) {   // 768 chunks each
      async_g2l_b128(ksb + (unsigned)i * 16, kg + (size_t)i * 8);
      async_g2l_b128(vsb + (unsigned)i * 16, vg + (size_t)i * 8);
    }
  }

  // Q fragments straight from global (bf16, A-operand layout) while async runs
  const int qrow = qc * CHUNK + wave * 16 + la;
  const u16* qp = Q + ((size_t)b * SQ + qrow) * DM + h * DH;
  bf16x16 qf[2];
  for (int ks = 0; ks < 2; ++ks) qf[ks] = frag_contig(qp + ks * 32 + half * 16);
  wait_async0();
  __syncthreads();

  // S = Q K^T * (1/sqrt(64)); 6 key tiles of 16
  float* Sw = Sb + (size_t)(wave * 16) * SKV;
  for (int jt = 0; jt < 6; ++jt) {
    f32x8 s; for (int e = 0; e < 8; ++e) s[e] = 0.f;
    for (int ks = 0; ks < 2; ++ks) {
      bf16x16 kf = frag_contig(&Ks[(jt * 16 + la) * DH + ks * 32 + half * 16]);
      s = wmma_bf16(qf[ks], kf, s);
    }
    for (int r = 0; r < 8; ++r)
      Sw[(size_t)(half * 8 + r) * SKV + jt * 16 + la] = s[r] * 0.125f;
  }
  __syncthreads();

  // softmax per row (threads 0..63), write P as bf16 packed into same LDS
  if (tid < CHUNK) {
    float* row  = Sb + (size_t)tid * SKV;
    u16*   prow = (u16*)row;                   // bf16 row, stride 2*SKV u16
    float mx = -1e30f;
    for (int j = 0; j < SVALID; ++j) mx = fmaxf(mx, row[j]);
    float sum = 0.f;
    for (int j = 0; j < SVALID; ++j) { float e = __expf(row[j] - mx); row[j] = e; sum += e; }
    float inv = 1.f / sum;
    for (int j = 0; j < SVALID; ++j) prow[j] = f2bf(row[j] * inv);
    for (int j = SVALID; j < SKV; ++j) prow[j] = 0;
  }
  __syncthreads();

  // P fragments (A operand, key-dim = 96 -> 3 k-steps); P row stride = 2*SKV u16
  const u16* Pw = (const u16*)Sb + (size_t)(wave * 16) * (2 * SKV);
  bf16x16 pf[3];
  for (int ks = 0; ks < 3; ++ks)
    pf[ks] = frag_contig(Pw + (size_t)la * (2 * SKV) + ks * 32 + half * 16);

  // O = P V ; 4 dh tiles of 16
  u16* orow = O + ((size_t)b * SQ + qc * CHUNK + wave * 16) * DM + h * DH;
  for (int nt = 0; nt < 4; ++nt) {
    f32x8 o; for (int e = 0; e < 8; ++e) o[e] = 0.f;
    for (int ks = 0; ks < 3; ++ks) {
      bf16x16 vf = frag_contig(&Vs[(nt * 16 + la) * SKV + ks * 32 + half * 16]);
      o = wmma_bf16(pf[ks], vf, o);
    }
    for (int r = 0; r < 8; ++r)
      orow[(size_t)(half * 8 + r) * DM + nt * 16 + la] = f2bf(o[r]);
  }
}

// ---------------------------------------------------------------------------
// Launch
// ---------------------------------------------------------------------------
extern "C" void kernel_launch(void* const* d_in, const int* in_sizes, int n_in,
                              void* d_out, int out_size, void* d_ws, size_t ws_size,
                              hipStream_t stream) {
  (void)in_sizes; (void)n_in; (void)out_size; (void)ws_size;
  const float* hs  = (const float*)d_in[0];
  const float* enc = (const float*)d_in[1];
  const float* sty = (const float*)d_in[2];
  const float* Wq  = (const float*)d_in[3];
  const float* Wk  = (const float*)d_in[4];
  const float* Wv  = (const float*)d_in[5];
  const float* Wks = (const float*)d_in[6];
  const float* Wvs = (const float*)d_in[7];
  const float* Wo  = (const float*)d_in[8];
  const float* bo  = (const float*)d_in[9];

  constexpr size_t QBYTES = (size_t)NB * SQ * DM * 2;        // 83,886,080
  constexpr size_t KBYTES = (size_t)NB * NH * SKV * DH * 2;  //  1,966,080
  constexpr size_t WBYTES = (size_t)DM * DM * 2;             //  3,276,800
  char* ws  = (char*)d_ws;
  u16*  Qb  = (u16*)(ws);
  u16*  Ob  = (u16*)(ws + QBYTES);
  u16*  Kc  = (u16*)(ws + 2 * QBYTES);
  u16*  Vtp = (u16*)(ws + 2 * QBYTES + KBYTES);
  u16*  Wqt = (u16*)(ws + 2 * QBYTES + 2 * KBYTES);
  u16*  Wot = (u16*)(ws + 2 * QBYTES + 2 * KBYTES + WBYTES);

  const int M = NB * SQ;  // 32768

  // 1) K/V (+style) projections -> padded bf16 K and V^T
  kv_prep_kernel<<<(NB * SKV * DM + 255) / 256, 256, 0, stream>>>(
      enc, sty, Wk, Wv, Wks, Wvs, Kc, Vtp);

  // 2) one-time weight convert+transpose (bf16, [N,K])
  wt_prep_kernel<<<(DM * DM + 255) / 256, 256, 0, stream>>>(Wq, Wqt, DM, DM);
  wt_prep_kernel<<<(DM * DM + 255) / 256, 256, 0, stream>>>(Wo, Wot, DM, DM);

  // 3) Q = hidden @ Wq  (f32 A, bf16 out)
  gemm_wmma_kernel<0, 1, 0><<<dim3(DM / 128, M / 128), 256, 0, stream>>>(
      hs, Wqt, Qb, nullptr, M, DM, DM);

  // 4) attention core
  attn_wmma_kernel<<<NB * NH * (SQ / 64), 128, 0, stream>>>(Qb, Kc, Vtp, Ob);

  // 5) out = O @ Wo + bo  (bf16 A, f32 out, bias)
  gemm_wmma_kernel<1, 0, 1><<<dim3(DM / 128, M / 128), 256, 0, stream>>>(
      Ob, Wot, (float*)d_out, bo, M, DM, DM);
}
